// AlexNet_Militrary_79130477462203
// MI455X (gfx1250) — matrix operand order repaired and checked by reading, not compiled, not run on hardware
//
#include <hip/hip_runtime.h>
#include <hip/hip_bf16.h>
#include <math.h>

typedef __attribute__((ext_vector_type(16))) _Float16 v16h;
typedef __attribute__((ext_vector_type(8)))  _Float16 v8h;
typedef __attribute__((ext_vector_type(8)))  float    v8f;

// ---------------------------------------------------------------------------
// Tiling: 128 threads = 4 waves (wave32). Block tile 64(M) x 128(N) x 32(K).
// Waves in 2x2: wave w -> (mw, nw) = (w&1, w>>1); each wave computes 32x64
// via 2 A frags x 4 B frags = 8 WMMAs per K-step. Double-buffered LDS.
// Layouts: activations NHWC f16, weights [Cout][kh][kw][ci] f16 (K padded).
// ---------------------------------------------------------------------------
#define BM 64
#define BN 128
#define BK 32
#define LDP 40     // LDS row pitch in halfs (80B, keeps b128 reads 16B aligned)

#define MODE_GENERIC 0   // scalar im2col gather (conv1, Cin=3)
#define MODE_ALIGNED 1   // Cin % 32 == 0: each K-step = 32 contiguous channels
#define MODE_FC      2   // KH=KW=Hin=Win=1: pure GEMM

struct ConvParams {
    const _Float16* in;    // [B, Hin, Win, Cin] f16 (NHWC)
    const _Float16* wgt;   // [Cout, Kp] f16, K ordered (kh, kw, ci), zero padded
    const float*    bias;  // [Cout] f32
    _Float16*       out;   // [B, Ho, Wo, Cout] f16 (ReLU applied)
    int B, Cin, Hin, Win;
    int Cout, KH, KW, stride, pad;
    int Ho, Wo;
    int K, Kp;
    int mode, cinSteps;    // cinSteps = Cin/32 (aligned mode)
    unsigned mHoWo, mWo;   // magic multipliers: q = umulhi(n, m) == n / d
};

// ---------------------------------------------------------------------------
// gfx1250 async global->LDS copy (ASYNCcnt path). LDS byte offset is the low
// 32 bits of the generic shared pointer (ISA 10.2 aperture truncation).
// ---------------------------------------------------------------------------
__device__ __forceinline__ void async_load_b128(void* lds_dst, const void* gsrc) {
    unsigned l = (unsigned)(uintptr_t)lds_dst;
    asm volatile("global_load_async_to_lds_b128 %0, %1, off"
                 :: "v"(l), "v"(gsrc) : "memory");
}
__device__ __forceinline__ void wait_async() {
    asm volatile("s_wait_asynccnt 0x0" ::: "memory");
}

__global__ __launch_bounds__(128) void conv_gemm_wmma(ConvParams p) {
    __shared__ _Float16 As[2][BM * LDP];   // As[buf][row][k]  (weights)
    __shared__ _Float16 Bs[2][BN * LDP];   // Bs[buf][col][k]  (im2col, K-contig)

    const int t    = threadIdx.x;
    const int wave = t >> 5;
    const int lane = t & 31;
    const int mw   = wave & 1;
    const int nw   = wave >> 1;

    const int nBase = blockIdx.x * BN;
    const int mBase = blockIdx.y * BM;
    const int HoWo  = p.Ho * p.Wo;
    const int Ntot  = p.B * HoWo;

    // ---- per-thread column (output pixel) state, hoisted out of K loop ----
    const int  pix = nBase + t;
    const bool pv  = pix < Ntot;
    int bb, oh, ow;
    if (p.mode == MODE_FC) { bb = pix; oh = 0; ow = 0; }
    else {
        bb = (int)__umulhi((unsigned)pix, p.mHoWo);
        int rem = pix - bb * HoWo;
        oh = (int)__umulhi((unsigned)rem, p.mWo);
        ow = rem - oh * p.Wo;
    }
    const int    ih0    = oh * p.stride - p.pad;
    const int    iw0    = ow * p.stride - p.pad;
    const size_t inRowB = (size_t)bb * p.Hin * p.Win * p.Cin;   // NHWC batch base

    // incremental K decomposition state (K order = kh, kw, ci)
    int g_ci = 0, g_kw = 0, g_kh = 0;        // generic: per element
    int a_cc = 0, a_kw = 0, a_kh = 0;        // aligned: per K-step (ci0 = a_cc*32)

    v8f acc[2][4] = {};
    const int kSteps = p.Kp / BK;

    // ---- stage K-step `ks` into LDS buffer `buf` (asyncs not yet waited) ----
    auto stage = [&](int ks, int buf) {
        const int kBase = ks * BK;
        // A tile: 64x32 weights, 2 x async b128 per thread; always in-bounds
        // (Cout multiple of 64, Kp multiple of 32 in every launch).
        #pragma unroll
        for (int j = 0; j < 2; ++j) {
            int c    = t * 2 + j;
            int row  = c >> 2;
            int kOff = (c & 3) * 8;
            const _Float16* src = p.wgt + (size_t)(mBase + row) * p.Kp + kBase + kOff;
            __builtin_prefetch(src + BK, 0, 0);          // global_prefetch_b8
            async_load_b128(&As[buf][row * LDP + kOff], src);
        }
        // B tile: col = t, k = kBase..kBase+31
        if (p.mode == MODE_FC) {
            if (pv) {
                const _Float16* src = p.in + (size_t)pix * p.Cin + kBase;
                #pragma unroll
                for (int g = 0; g < 4; ++g)
                    async_load_b128(&Bs[buf][t * LDP + g * 8], src + g * 8);
            } else {
                v8h z = {};
                #pragma unroll
                for (int g = 0; g < 4; ++g)
                    *(v8h*)(&Bs[buf][t * LDP + g * 8]) = z;
            }
        } else if (p.mode == MODE_ALIGNED) {
            // one (kh,kw) position, 32 contiguous channels
            const int ih = ih0 + a_kh;
            const int iw = iw0 + a_kw;
            const int ci0 = a_cc * 32;
            if (pv && (unsigned)ih < (unsigned)p.Hin && (unsigned)iw < (unsigned)p.Win) {
                const _Float16* src =
                    p.in + inRowB + ((size_t)ih * p.Win + iw) * p.Cin + ci0;
                #pragma unroll
                for (int g = 0; g < 4; ++g)
                    async_load_b128(&Bs[buf][t * LDP + g * 8], src + g * 8);
            } else {
                v8h z = {};
                #pragma unroll
                for (int g = 0; g < 4; ++g)
                    *(v8h*)(&Bs[buf][t * LDP + g * 8]) = z;
            }
            ++a_cc;
            if (a_cc == p.cinSteps) { a_cc = 0; ++a_kw; if (a_kw == p.KW) { a_kw = 0; ++a_kh; } }
        } else {
            // generic scalar gather (conv1 only: Cin=3, K=27->Kp=32)
            int kk = kBase;
            #pragma unroll
            for (int g = 0; g < 4; ++g) {
                v8h sv;
                #pragma unroll
                for (int e = 0; e < 8; ++e) {
                    _Float16 v = (_Float16)0.f;
                    int ih = ih0 + g_kh;
                    int iw = iw0 + g_kw;
                    if (pv && kk < p.K &&
                        (unsigned)ih < (unsigned)p.Hin &&
                        (unsigned)iw < (unsigned)p.Win)
                        v = p.in[inRowB + ((size_t)ih * p.Win + iw) * p.Cin + g_ci];
                    sv[e] = v;
                    ++kk;
                    ++g_ci;
                    if (g_ci == p.Cin) { g_ci = 0; ++g_kw; if (g_kw == p.KW) { g_kw = 0; ++g_kh; } }
                }
                *(v8h*)(&Bs[buf][t * LDP + g * 8]) = sv;
            }
        }
    };

    stage(0, 0);
    int cur = 0;
    for (int ks = 0; ks < kSteps; ++ks) {
        wait_async();        // buffer `cur` fully landed in LDS
        __syncthreads();     // visible to all waves; prev reads of cur^1 done
        if (ks + 1 < kSteps) stage(ks + 1, cur ^ 1);   // overlap with compute

        // ---- fragments (ISA 7.12.2 layouts) + 8 WMMAs ----
        const int lm  = lane & 15;
        const int ak0 = (lane < 16) ? 0 : 8;
        v16h af[2];
        #pragma unroll
        for (int r = 0; r < 2; ++r) {
            const _Float16* ap = &As[cur][(mw * 32 + r * 16 + lm) * LDP + ak0];
            v8h alo = *(const v8h*)(ap);
            v8h ahi = *(const v8h*)(ap + 16);
            af[r] = __builtin_shufflevector(alo, ahi,
                0,1,2,3,4,5,6,7, 8,9,10,11,12,13,14,15);
        }
        const int bk0 = (lane < 16) ? 0 : 8;
        #pragma unroll
        for (int ng = 0; ng < 4; ++ng) {
            const _Float16* bp = &Bs[cur][(nw * 64 + ng * 16 + lm) * LDP + bk0];
            v8h blo = *(const v8h*)(bp);
            v8h bhi = *(const v8h*)(bp + 16);
            v16h bf = __builtin_shufflevector(blo, bhi,
                0,1,2,3,4,5,6,7, 8,9,10,11,12,13,14,15);
            acc[0][ng] = __builtin_amdgcn_wmma_f32_16x16x32_f16(
                false, af[0], false, bf, (short)0, acc[0][ng], false, false);
            acc[1][ng] = __builtin_amdgcn_wmma_f32_16x16x32_f16(
                false, af[1], false, bf, (short)0, acc[1][ng], false, false);
        }
        cur ^= 1;
    }

    // ---- epilogue: bias + ReLU -> NHWC f16, vectorized b128 stores ----
    const int rowOff = (lane >= 16) ? 8 : 0;
    const int lm     = lane & 15;
    #pragma unroll
    for (int ng = 0; ng < 4; ++ng) {
        int opix = nBase + nw * 64 + ng * 16 + lm;
        if (opix >= Ntot) continue;
        // NHWC: out offset = opix * Cout + m (pixel-major, channels contiguous)
        _Float16* obase = p.out + (size_t)opix * p.Cout;
        #pragma unroll
        for (int r = 0; r < 2; ++r) {
            int m0 = mBase + mw * 32 + r * 16 + rowOff;
            v8h ov;
            #pragma unroll
            for (int i = 0; i < 8; ++i) {
                float v = acc[r][ng][i] + p.bias[m0 + i];
                ov[i] = (_Float16)fmaxf(v, 0.f);
            }
            *(v8h*)(obase + m0) = ov;   // 16B store, m0 multiple of 8, Cout of 64
        }
    }
}

// ---------------------------------------------------------------------------
// NCHW f32 -> NHWC f16 (input image transpose+convert)
// ---------------------------------------------------------------------------
__global__ void nchw_to_nhwc_f16(const float* __restrict__ in, _Float16* __restrict__ out,
                                 int C, int H, int W, int total) {
    int i = blockIdx.x * blockDim.x + threadIdx.x;
    if (i >= total) return;
    int c = i % C;
    int t2 = i / C;
    int w = t2 % W; t2 /= W;
    int h = t2 % H;
    int b = t2 / H;
    out[i] = (_Float16)in[(((size_t)b * C + c) * H + h) * W + w];
}

// ---------------------------------------------------------------------------
// Weight convert: OIHW f32 -> [O][kh][kw][ci] f16, K zero-padded to Kp.
// Also handles fc1's (c,h,w)->(h,w,c) reorder and fc2 identity (KH=KW=1).
// ---------------------------------------------------------------------------
__global__ void wconvert_nhwc(const float* __restrict__ in, _Float16* __restrict__ out,
                              int Cin, int KH, int KW, int K, int Kp, int total) {
    int i = blockIdx.x * blockDim.x + threadIdx.x;
    if (i >= total) return;
    int o  = i / Kp;
    int kp = i - o * Kp;
    _Float16 v = (_Float16)0.f;
    if (kp < K) {
        int ci = kp % Cin;
        int r  = kp / Cin;
        int kw = r % KW;
        int kh = r / KW;
        v = (_Float16)in[((size_t)o * Cin + ci) * (KH * KW) + kh * KW + kw];
    }
    out[i] = v;
}

// ---------------------------------------------------------------------------
// 3x3 stride-2 VALID max pool, NHWC f16 (channels contiguous -> coalesced)
// ---------------------------------------------------------------------------
__global__ void maxpool3s2_nhwc(const _Float16* __restrict__ in, _Float16* __restrict__ out,
                                int C, int H, int W, int Ho, int Wo, int total) {
    int i = blockIdx.x * blockDim.x + threadIdx.x;
    if (i >= total) return;
    int c  = i % C;
    int t2 = i / C;
    int wo = t2 % Wo; t2 /= Wo;
    int ho = t2 % Ho;
    int b  = t2 / Ho;
    int h0 = ho * 2, w0 = wo * 2;
    float m = -INFINITY;
    #pragma unroll
    for (int kh = 0; kh < 3; ++kh)
        #pragma unroll
        for (int kw = 0; kw < 3; ++kw) {
            int h = h0 + kh, w = w0 + kw;
            if (h < H && w < W)
                m = fmaxf(m, (float)in[(((size_t)b * H + h) * W + w) * C + c]);
        }
    out[i] = (_Float16)m;
}

// ---------------------------------------------------------------------------
// fc3 (10 x 4096) + argmax-fill + noise + softmax. One block per batch row.
// ---------------------------------------------------------------------------
__global__ __launch_bounds__(256) void fc3_post(const _Float16* __restrict__ act,
                                                const float* __restrict__ w,
                                                const float* __restrict__ bias,
                                                const float* __restrict__ unif,
                                                const float* __restrict__ noise,
                                                float* __restrict__ out) {
    const int batch = blockIdx.x;
    const int t = threadIdx.x;
    __shared__ float red[256];
    __shared__ float logits[10];

    for (int o = 0; o < 10; ++o) {
        float s = 0.f;
        for (int k = t; k < 4096; k += 256)
            s += (float)act[(size_t)batch * 4096 + k] * w[(size_t)o * 4096 + k];
        red[t] = s;
        __syncthreads();
        for (int off = 128; off > 0; off >>= 1) {
            if (t < off) red[t] += red[t + off];
            __syncthreads();
        }
        if (t == 0) logits[o] = red[0] + bias[o];
        __syncthreads();
    }

    if (t == 0) {
        float m = logits[0];
        int idx = 0;
        for (int j = 1; j < 10; ++j)
            if (logits[j] > m) { m = logits[j]; idx = j; }
        float z[10];
        float zmax = -INFINITY;
        for (int j = 0; j < 10; ++j) {
            float f = (j == idx) ? logits[j] : unif[batch * 10 + j] * m;
            z[j] = f + noise[batch * 10 + j];
            zmax = fmaxf(zmax, z[j]);
        }
        float se = 0.f;
        for (int j = 0; j < 10; ++j) { z[j] = expf(z[j] - zmax); se += z[j]; }
        float inv = 1.f / se;
        for (int j = 0; j < 10; ++j) out[batch * 10 + j] = z[j] * inv;
    }
}

// ---------------------------------------------------------------------------
// Host orchestration
// ---------------------------------------------------------------------------
static inline unsigned cdiv(long long a, long long b) { return (unsigned)((a + b - 1) / b); }
static inline unsigned magic_u32(unsigned d) {
    // exact q = umulhi(n, magic) for n < 2^20, 1 < d < 2^12 (our ranges)
    return d <= 1 ? 0u : (unsigned)(0xFFFFFFFFull / d + 1ull);
}

extern "C" void kernel_launch(void* const* d_in, const int* in_sizes, int n_in,
                              void* d_out, int out_size, void* d_ws, size_t ws_size,
                              hipStream_t stream) {
    (void)in_sizes; (void)n_in; (void)out_size; (void)ws_size;

    const float* x     = (const float*)d_in[0];
    const float* w1    = (const float*)d_in[1];
    const float* b1    = (const float*)d_in[2];
    const float* w2    = (const float*)d_in[3];
    const float* b2    = (const float*)d_in[4];
    const float* w3    = (const float*)d_in[5];
    const float* b3    = (const float*)d_in[6];
    const float* w4    = (const float*)d_in[7];
    const float* b4    = (const float*)d_in[8];
    const float* w5    = (const float*)d_in[9];
    const float* b5    = (const float*)d_in[10];
    const float* fc1_w = (const float*)d_in[11];
    const float* fc1_b = (const float*)d_in[12];
    const float* fc2_w = (const float*)d_in[13];
    const float* fc2_b = (const float*)d_in[14];
    const float* fc3_w = (const float*)d_in[15];
    const float* fc3_b = (const float*)d_in[16];
    const float* unif  = (const float*)d_in[17];
    const float* noise = (const float*)d_in[18];
    float* out = (float*)d_out;

    const int B = 128;

    char* ws = (char*)d_ws;
    size_t off = 0;
    auto take = [&](size_t halfs) -> _Float16* {
        _Float16* p = (_Float16*)(ws + off);
        off += (halfs * sizeof(_Float16) + 255) & ~(size_t)255;
        return p;
    };
    _Float16* x16  = take((size_t)B * 224 * 224 * 3);       // NHWC
    _Float16* w1h  = take((size_t)64 * 32);
    _Float16* w2h  = take((size_t)192 * 1600);
    _Float16* w3h  = take((size_t)384 * 1728);
    _Float16* w4h  = take((size_t)256 * 3456);
    _Float16* w5h  = take((size_t)256 * 2304);
    _Float16* fc1h = take((size_t)4096 * 9216);
    _Float16* fc2h = take((size_t)4096 * 4096);
    _Float16* a1   = take((size_t)B * 57 * 57 * 64);
    _Float16* p1   = take((size_t)B * 28 * 28 * 64);
    _Float16* a2   = take((size_t)B * 28 * 28 * 192);
    _Float16* p2   = take((size_t)B * 13 * 13 * 192);
    _Float16* a3   = take((size_t)B * 13 * 13 * 384);
    _Float16* a4   = take((size_t)B * 13 * 13 * 256);
    _Float16* a5   = take((size_t)B * 13 * 13 * 256);
    _Float16* p5   = take((size_t)B * 6 * 6 * 256);
    _Float16* f1   = take((size_t)B * 4096);
    _Float16* f2   = take((size_t)B * 4096);

    {
        int n = B * 3 * 224 * 224;
        nchw_to_nhwc_f16<<<cdiv(n, 256), 256, 0, stream>>>(x, x16, 3, 224, 224, n);
    }
    // weights -> f16, K order (kh, kw, ci)
    wconvert_nhwc<<<cdiv(64   * 32,   256), 256, 0, stream>>>(w1,    w1h,  3,    3, 3, 27,   32,   64 * 32);
    wconvert_nhwc<<<cdiv(192  * 1600, 256), 256, 0, stream>>>(w2,    w2h,  64,   5, 5, 1600, 1600, 192 * 1600);
    wconvert_nhwc<<<cdiv(384  * 1728, 256), 256, 0, stream>>>(w3,    w3h,  192,  3, 3, 1728, 1728, 384 * 1728);
    wconvert_nhwc<<<cdiv(256  * 3456, 256), 256, 0, stream>>>(w4,    w4h,  384,  3, 3, 3456, 3456, 256 * 3456);
    wconvert_nhwc<<<cdiv(256  * 2304, 256), 256, 0, stream>>>(w5,    w5h,  256,  3, 3, 2304, 2304, 256 * 2304);
    // fc1: reorder (c,h,w) -> (h,w,c) so it consumes the NHWC pool5 flatten
    wconvert_nhwc<<<cdiv((long long)4096 * 9216, 256), 256, 0, stream>>>(fc1_w, fc1h, 256, 6, 6, 9216, 9216, (int)(4096 * 9216));
    // fc2: identity reorder (KH=KW=1)
    wconvert_nhwc<<<cdiv((long long)4096 * 4096, 256), 256, 0, stream>>>(fc2_w, fc2h, 4096, 1, 1, 4096, 4096, (int)(4096 * 4096));

    auto conv = [&](const _Float16* in, const _Float16* wgt, const float* bias,
                    _Float16* o, int Cin, int Hin, int Win, int Cout,
                    int KH, int KW, int stride, int pad, int Ho, int Wo,
                    int K, int Kp) {
        ConvParams cp;
        cp.in = in; cp.wgt = wgt; cp.bias = bias; cp.out = o;
        cp.B = B; cp.Cin = Cin; cp.Hin = Hin; cp.Win = Win;
        cp.Cout = Cout; cp.KH = KH; cp.KW = KW; cp.stride = stride; cp.pad = pad;
        cp.Ho = Ho; cp.Wo = Wo; cp.K = K; cp.Kp = Kp;
        if (KH == 1 && KW == 1 && Hin == 1 && Win == 1) cp.mode = MODE_FC;
        else if ((Cin & 31) == 0)                       cp.mode = MODE_ALIGNED;
        else                                            cp.mode = MODE_GENERIC;
        cp.cinSteps = Cin / 32;
        cp.mHoWo = magic_u32((unsigned)(Ho * Wo));
        cp.mWo   = magic_u32((unsigned)Wo);
        long long Ntot = (long long)B * Ho * Wo;
        dim3 grid(cdiv(Ntot, BN), cdiv(Cout, BM));
        conv_gemm_wmma<<<grid, 128, 0, stream>>>(cp);
    };

    // conv1: [B,224,224,3] -> [B,57,57,64], 3x3 s4 p2 (K=27 -> Kp=32, generic)
    conv(x16, w1h, b1, a1, 3, 224, 224, 64, 3, 3, 4, 2, 57, 57, 27, 32);
    {
        int total = B * 28 * 28 * 64;
        maxpool3s2_nhwc<<<cdiv(total, 256), 256, 0, stream>>>(a1, p1, 64, 57, 57, 28, 28, total);
    }
    // conv2: 5x5 s1 p2 (aligned: Cin=64)
    conv(p1, w2h, b2, a2, 64, 28, 28, 192, 5, 5, 1, 2, 28, 28, 1600, 1600);
    {
        int total = B * 13 * 13 * 192;
        maxpool3s2_nhwc<<<cdiv(total, 256), 256, 0, stream>>>(a2, p2, 192, 28, 28, 13, 13, total);
    }
    conv(p2, w3h, b3, a3, 192, 13, 13, 384, 3, 3, 1, 1, 13, 13, 1728, 1728);
    conv(a3, w4h, b4, a4, 384, 13, 13, 256, 3, 3, 1, 1, 13, 13, 3456, 3456);
    conv(a4, w5h, b5, a5, 256, 13, 13, 256, 3, 3, 1, 1, 13, 13, 2304, 2304);
    {
        int total = B * 6 * 6 * 256;
        maxpool3s2_nhwc<<<cdiv(total, 256), 256, 0, stream>>>(a5, p5, 256, 13, 13, 6, 6, total);
    }
    // fc1 / fc2 as pure GEMM (fc1 weights pre-reordered to NHWC flatten order)
    conv(p5, fc1h, fc1_b, f1, 9216, 1, 1, 4096, 1, 1, 1, 0, 1, 1, 9216, 9216);
    conv(f1, fc2h, fc2_b, f2, 4096, 1, 1, 4096, 1, 1, 1, 0, 1, 1, 4096, 4096);
    fc3_post<<<B, 256, 0, stream>>>(f2, fc3_w, fc3_b, unif, noise, out);
}
